// Custom_Self_Attention_28621662060731
// MI455X (gfx1250) — matrix operand order repaired and checked by reading, not compiled, not run on hardware
//
#include <hip/hip_runtime.h>
#include <math.h>

#define IN_DIM 128
#define ATTN   64
#define NCLASS 4
#define HW     16384
#define S1     4096
#define S2     4096
#define KW     384   // 3 taps * 128 channels, K reindexed as k' = kh*128 + i

typedef _Float16 f16;
typedef __attribute__((ext_vector_type(4)))  float    v4f;
typedef __attribute__((ext_vector_type(8)))  float    v8f;
typedef __attribute__((ext_vector_type(8)))  _Float16 v8h;
typedef __attribute__((ext_vector_type(16))) _Float16 v16h;

__device__ __forceinline__ float sigmoidf_(float x) {
    return 1.0f / (1.0f + expf(-x));
}

__device__ __forceinline__ v16h cat8(v8h p0, v8h p1) {
    return __builtin_shufflevector(p0, p1, 0,1,2,3,4,5,6,7,8,9,10,11,12,13,14,15);
}

// ---------------------------------------------------------------------------
// Repack OIHW 3x3 weights: keep center column (kw==1), reindex K as
// k' = kh*128 + i, split into hi/lo f16, row-major [a][384].
// ---------------------------------------------------------------------------
__global__ void repack_weights_kernel(const float* __restrict__ w,
                                      f16* __restrict__ Wh,
                                      f16* __restrict__ Wl) {
    int t = blockIdx.x * blockDim.x + threadIdx.x;
    if (t >= ATTN * KW) return;
    int a = t / KW;
    int k = t % KW;
    int kh = k >> 7;             // tap
    int i  = k & 127;            // input channel
    float val = w[((a * IN_DIM + i) * 3 + kh) * 3 + 1];
    f16 h = (f16)val;
    Wh[t] = h;
    Wl[t] = (f16)(val - (float)h);
}

// ---------------------------------------------------------------------------
// Implicit-GEMM 1-D conv (K=3, pad 1, 128->64 ch, both weight sets) on
// V_WMMA_F32_16X16X32_F16 with hi/lo f16 split (f32-grade accuracy), fused
// bias + per-position L2 normalization.
//
// Block: 256 thr = 8 waves, 128 positions. Wave: 16 positions x 64 channels
// (4 M-tiles x 2 sets = 8 accumulators).  B fragment (X[k',p]) is built from
// the LDS-staged input tile: chunk c -> tap kh=c/4, 32 input
// channels (half h holds 8h..8h+7, 16+8h..23+8h), converted to hi/lo f16 once and reused by 24 WMMAs.
// ---------------------------------------------------------------------------
__global__ void conv_wmma_kernel(const float* __restrict__ in, int L,
                                 const f16* __restrict__ W1h, const f16* __restrict__ W1l,
                                 const f16* __restrict__ W2h, const f16* __restrict__ W2l,
                                 const float* __restrict__ b1, const float* __restrict__ b2,
                                 float* __restrict__ out1, float* __restrict__ out2,
                                 f16* __restrict__ h1, f16* __restrict__ l1,
                                 f16* __restrict__ h2, f16* __restrict__ l2) {
    __shared__ float sTile[IN_DIM][131];      // input cols p0-1 .. p0+128 (130 used)
    __shared__ float sB[2 * ATTN];            // biases
    const int t  = threadIdx.x;               // 0..255
    const int p0 = blockIdx.x * 128;

    for (int idx = t; idx < IN_DIM * 130; idx += 256) {
        int i = idx / 130, j = idx % 130;
        int p = p0 - 1 + j;
        sTile[i][j] = (p >= 0 && p < L) ? in[i * L + p] : 0.0f;
    }
    if (t < ATTN)            sB[t]       = b1[t];
    else if (t < 2 * ATTN)   sB[t]       = b2[t - ATTN];
    __syncthreads();

    const int lane     = t & 31;
    const int wid      = t >> 5;
    const int lrow     = lane & 15;
    const int halfSel  = lane >> 4;
    const int posLocal = wid * 16 + lrow;     // this lane's position (0..127)

    v8f acc1[4] = {};
    v8f acc2[4] = {};

    #pragma unroll
    for (int c = 0; c < 12; ++c) {            // K chunks of 32 (K' = kh*128+i)
        const int kh = c >> 2;
        const int i0 = (c & 3) * 32 + halfSel * 8;
        const int jc = posLocal + kh;

        // build B fragment (hi/lo) from staged f32 input
        v16h bh, bl;
        #pragma unroll
        for (int e = 0; e < 16; ++e) {
            float x = sTile[i0 + e + ((e >> 3) << 3)][jc];
            f16 hx = (f16)x;
            bh[e] = hx;
            bl[e] = (f16)(x - (float)hx);
        }

        const int ko = c * 32 + halfSel * 8;  // A-fragment K base for this lane
        #pragma unroll
        for (int m = 0; m < 4; ++m) {
            const f16* r1h = W1h + (m * 16 + lrow) * KW + ko;
            const f16* r1l = W1l + (m * 16 + lrow) * KW + ko;
            const f16* r2h = W2h + (m * 16 + lrow) * KW + ko;
            const f16* r2l = W2l + (m * 16 + lrow) * KW + ko;
            v16h a1h = cat8(*(const v8h*)(r1h), *(const v8h*)(r1h + 16));
            v16h a1l = cat8(*(const v8h*)(r1l), *(const v8h*)(r1l + 16));
            v16h a2h = cat8(*(const v8h*)(r2h), *(const v8h*)(r2h + 16));
            v16h a2l = cat8(*(const v8h*)(r2l), *(const v8h*)(r2l + 16));
            acc1[m] = __builtin_amdgcn_wmma_f32_16x16x32_f16(
                false, a1h, false, bh, (short)0, acc1[m], false, false);
            acc1[m] = __builtin_amdgcn_wmma_f32_16x16x32_f16(
                false, a1h, false, bl, (short)0, acc1[m], false, false);
            acc1[m] = __builtin_amdgcn_wmma_f32_16x16x32_f16(
                false, a1l, false, bh, (short)0, acc1[m], false, false);
            acc2[m] = __builtin_amdgcn_wmma_f32_16x16x32_f16(
                false, a2h, false, bh, (short)0, acc2[m], false, false);
            acc2[m] = __builtin_amdgcn_wmma_f32_16x16x32_f16(
                false, a2h, false, bl, (short)0, acc2[m], false, false);
            acc2[m] = __builtin_amdgcn_wmma_f32_16x16x32_f16(
                false, a2l, false, bh, (short)0, acc2[m], false, false);
        }
    }

    // bias + per-position sum of squares (this lane holds 32 of 64 channels;
    // the partner half-lane holds the other 32)
    float ss1 = 0.0f, ss2 = 0.0f;
    #pragma unroll
    for (int m = 0; m < 4; ++m) {
        const int chBase = m * 16 + halfSel * 8;
        #pragma unroll
        for (int r = 0; r < 8; ++r) {
            float v1 = acc1[m][r] + sB[chBase + r];
            float v2 = acc2[m][r] + sB[ATTN + chBase + r];
            acc1[m][r] = v1;
            acc2[m][r] = v2;
            ss1 = fmaf(v1, v1, ss1);
            ss2 = fmaf(v2, v2, ss2);
        }
    }
    ss1 += __shfl_xor(ss1, 16);
    ss2 += __shfl_xor(ss2, 16);
    const float inv1 = 1.0f / sqrtf(ss1);
    const float inv2 = 1.0f / sqrtf(ss2);

    const int p = p0 + posLocal;
    #pragma unroll
    for (int m = 0; m < 4; ++m) {
        const int chBase = m * 16 + halfSel * 8;
        float* o1 = out1 + p * 64 + chBase;
        float* o2 = out2 + p * 64 + chBase;
        v4f w1a, w1b, w2a, w2b;
        v8h ph1, pl1, ph2, pl2;
        #pragma unroll
        for (int r = 0; r < 8; ++r) {
            float n1 = acc1[m][r] * inv1;
            float n2 = acc2[m][r] * inv2;
            if (r < 4) { w1a[r] = n1; w2a[r] = n2; }
            else       { w1b[r - 4] = n1; w2b[r - 4] = n2; }
            f16 hi1 = (f16)n1, hi2 = (f16)n2;
            ph1[r] = hi1; pl1[r] = (f16)(n1 - (float)hi1);
            ph2[r] = hi2; pl2[r] = (f16)(n2 - (float)hi2);
        }
        *(v4f*)(o1)     = w1a;
        *(v4f*)(o1 + 4) = w1b;
        *(v4f*)(o2)     = w2a;
        *(v4f*)(o2 + 4) = w2b;
        if (h1) {
            *(v8h*)(h1 + p * 64 + chBase) = ph1;
            *(v8h*)(l1 + p * 64 + chBase) = pl1;
            *(v8h*)(h2 + p * 64 + chBase) = ph2;
            *(v8h*)(l2 + p * 64 + chBase) = pl2;
        }
    }
}

// ---------------------------------------------------------------------------
// Per-class mean of normalized k-vectors (deterministic, single block).
// ---------------------------------------------------------------------------
__global__ void class_reduce_kernel(const float* __restrict__ k1hat,
                                    const float* __restrict__ k2hat,
                                    const int* __restrict__ label,
                                    float* __restrict__ K1c,
                                    float* __restrict__ K2c) {
    int t  = threadIdx.x;                     // 0..255
    int c  = t >> 6;
    int ch = t & 63;
    float a1 = 0.0f, a2 = 0.0f;
    int cnt = 0;
    for (int s = 0; s < S1; ++s) {
        if (label[s] == c) {
            a1 += k1hat[s * 64 + ch];
            a2 += k2hat[s * 64 + ch];
            ++cnt;
        }
    }
    float inv = 1.0f / (float)cnt;            // every class occurs (labels start 0..3)
    K1c[t] = a1 * inv;
    K2c[t] = a2 * inv;
}

// ---------------------------------------------------------------------------
// memory_map: sigmoid(cw0 * (sk1hat^T k1hat) + cw1 * (sk2hat^T k2hat) + cb)
// Split-f16 GEMM on V_WMMA_F32_16X16X32_F16; wave computes 16(M) x 64(N),
// A fragments loaded once and reused across 4 N-tiles.
// ---------------------------------------------------------------------------
__global__ void memory_map_wmma_kernel(
        const f16* __restrict__ skh1, const f16* __restrict__ skl1,
        const f16* __restrict__ skh2, const f16* __restrict__ skl2,
        const f16* __restrict__ kh1,  const f16* __restrict__ kl1,
        const f16* __restrict__ kh2,  const f16* __restrict__ kl2,
        const float* __restrict__ cc1w, const float* __restrict__ cc1b,
        float* __restrict__ out) {
    const int lane    = threadIdx.x & 31;
    const int wid     = threadIdx.x >> 5;
    const int mw      = wid & 3;
    const int nw      = wid >> 2;
    const int m0      = blockIdx.y * 64 + mw * 16;       // S2 rows
    const int nBase   = blockIdx.x * 128 + nw * 64;      // S1 cols
    const int lrow    = lane & 15;
    const int halfSel = lane >> 4;                       // 0 or 1

    const f16* aRow1h = skh1 + (m0 + lrow) * 64;
    const f16* aRow1l = skl1 + (m0 + lrow) * 64;
    const f16* aRow2h = skh2 + (m0 + lrow) * 64;
    const f16* aRow2l = skl2 + (m0 + lrow) * 64;

    v16h a1h[2], a1l[2], a2h[2], a2l[2];
    #pragma unroll
    for (int c = 0; c < 2; ++c) {
        const int ko = c * 32 + halfSel * 16;
        a1h[c] = cat8(*(const v8h*)(aRow1h + ko), *(const v8h*)(aRow1h + ko + 8));
        a1l[c] = cat8(*(const v8h*)(aRow1l + ko), *(const v8h*)(aRow1l + ko + 8));
        a2h[c] = cat8(*(const v8h*)(aRow2h + ko), *(const v8h*)(aRow2h + ko + 8));
        a2l[c] = cat8(*(const v8h*)(aRow2l + ko), *(const v8h*)(aRow2l + ko + 8));
    }

    v8f acc1[4] = {};
    v8f acc2[4] = {};
    #pragma unroll
    for (int t = 0; t < 4; ++t) {
        const int n0 = nBase + t * 16;
        const f16* bCol1h = kh1 + (n0 + lrow) * 64;
        const f16* bCol1l = kl1 + (n0 + lrow) * 64;
        const f16* bCol2h = kh2 + (n0 + lrow) * 64;
        const f16* bCol2l = kl2 + (n0 + lrow) * 64;
        #pragma unroll
        for (int c = 0; c < 2; ++c) {
            const int bo = c * 32 + halfSel * 16;
            v16h bh = *(const v16h*)(bCol1h + bo);
            v16h bl = *(const v16h*)(bCol1l + bo);
            acc1[t] = __builtin_amdgcn_wmma_f32_16x16x32_f16(
                false, a1h[c], false, bh, (short)0, acc1[t], false, false);
            acc1[t] = __builtin_amdgcn_wmma_f32_16x16x32_f16(
                false, a1h[c], false, bl, (short)0, acc1[t], false, false);
            acc1[t] = __builtin_amdgcn_wmma_f32_16x16x32_f16(
                false, a1l[c], false, bh, (short)0, acc1[t], false, false);
            v16h dh = *(const v16h*)(bCol2h + bo);
            v16h dl = *(const v16h*)(bCol2l + bo);
            acc2[t] = __builtin_amdgcn_wmma_f32_16x16x32_f16(
                false, a2h[c], false, dh, (short)0, acc2[t], false, false);
            acc2[t] = __builtin_amdgcn_wmma_f32_16x16x32_f16(
                false, a2h[c], false, dl, (short)0, acc2[t], false, false);
            acc2[t] = __builtin_amdgcn_wmma_f32_16x16x32_f16(
                false, a2l[c], false, dh, (short)0, acc2[t], false, false);
        }
    }

    const float w0 = cc1w[0], w1 = cc1w[1], bb = cc1b[0];
    const int row_base = m0 + halfSel * 8;
    #pragma unroll
    for (int t = 0; t < 4; ++t) {
        const int col = nBase + t * 16 + lrow;
        #pragma unroll
        for (int r = 0; r < 8; ++r) {
            float v = fmaf(w0, acc1[t][r], fmaf(w1, acc2[t][r], bb));
            out[(row_base + r) * S1 + col] = sigmoidf_(v);
        }
    }
}

// ---------------------------------------------------------------------------
// result_map: r[c,p] = dot(Kc[c], qhat[p]);  out = sigmoid(cc2_w @ r + cc2_b)
// ---------------------------------------------------------------------------
__global__ void result_map_kernel(const float* __restrict__ q1hat,
                                  const float* __restrict__ q2hat,
                                  const float* __restrict__ K1c,
                                  const float* __restrict__ K2c,
                                  const float* __restrict__ cc2w,
                                  const float* __restrict__ cc2b,
                                  float* __restrict__ out) {
    __shared__ float sK[512];
    int t = threadIdx.x;                      // 0..255
    sK[t]       = K1c[t];
    sK[256 + t] = K2c[t];
    __syncthreads();

    int p = blockIdx.x * 256 + t;
    const float* q1p = q1hat + p * 64;
    const float* q2p = q2hat + p * 64;
    float r[8];
    #pragma unroll
    for (int c = 0; c < NCLASS; ++c) {
        float s1 = 0.0f, s2 = 0.0f;
        #pragma unroll 8
        for (int ch = 0; ch < 64; ++ch) {
            s1 = fmaf(sK[c * 64 + ch],       q1p[ch], s1);
            s2 = fmaf(sK[256 + c * 64 + ch], q2p[ch], s2);
        }
        r[c]     = s1;
        r[4 + c] = s2;
    }
    #pragma unroll
    for (int o = 0; o < NCLASS; ++o) {
        float s = cc2b[o];
        #pragma unroll
        for (int j = 0; j < 8; ++j) s = fmaf(cc2w[o * 8 + j], r[j], s);
        out[o * HW + p] = sigmoidf_(s);
    }
}

// ---------------------------------------------------------------------------
extern "C" void kernel_launch(void* const* d_in, const int* in_sizes, int n_in,
                              void* d_out, int out_size, void* d_ws, size_t ws_size,
                              hipStream_t stream) {
    (void)in_sizes; (void)n_in; (void)out_size; (void)ws_size;

    // setup_inputs() dict order
    const float* input       = (const float*)d_in[0];   // [128][16384]
    const float* class_input = (const float*)d_in[1];   // [128][4096]
    const float* saved_input = (const float*)d_in[2];   // [128][4096]
    const int*   class_label = (const int*)  d_in[3];   // [4096]
    /* saved_class_label d_in[4] unused by the forward pass */
    const float* w1   = (const float*)d_in[5];
    const float* b1   = (const float*)d_in[6];
    const float* w2   = (const float*)d_in[7];
    const float* b2   = (const float*)d_in[8];
    const float* cc1w = (const float*)d_in[9];
    const float* cc1b = (const float*)d_in[10];
    const float* cc2w = (const float*)d_in[11];
    const float* cc2b = (const float*)d_in[12];

    float* outf       = (float*)d_out;
    float* result_map = outf;                 // [4][16384]
    float* memory_map = outf + NCLASS * HW;   // [4096][4096]

    // workspace carve-up (f32 region first, then 32B-aligned f16 region)
    float* ws = (float*)d_ws;
    float* q1hat  = ws;                         // 16384*64
    float* q2hat  = q1hat  + HW * 64;
    float* k1hat  = q2hat  + HW * 64;           // 4096*64
    float* k2hat  = k1hat  + S1 * 64;
    float* sk1hat = k2hat  + S1 * 64;
    float* sk2hat = sk1hat + S2 * 64;
    float* K1c    = sk2hat + S2 * 64;           // 4*64
    float* K2c    = K1c + NCLASS * 64;

    f16* f16base = (f16*)(K2c + NCLASS * 64);   // float offset 3,146,240 -> 32B aligned
    f16* skh1 = f16base;                        // each 4096*64 halves
    f16* skl1 = skh1 + S2 * 64;
    f16* skh2 = skl1 + S2 * 64;
    f16* skl2 = skh2 + S2 * 64;
    f16* kh1  = skl2 + S2 * 64;
    f16* kl1  = kh1  + S1 * 64;
    f16* kh2  = kl1  + S1 * 64;
    f16* kl2  = kh2  + S1 * 64;
    f16* W1h  = kl2  + S1 * 64;                 // each 64*384 halves
    f16* W1l  = W1h + ATTN * KW;
    f16* W2h  = W1l + ATTN * KW;
    f16* W2l  = W2h + ATTN * KW;

    // 1) repack + hi/lo split weights (center column, tap-major K)
    {
        int n = ATTN * KW;
        repack_weights_kernel<<<(n + 255) / 256, 256, 0, stream>>>(w1, W1h, W1l);
        repack_weights_kernel<<<(n + 255) / 256, 256, 0, stream>>>(w2, W2h, W2l);
    }

    // 2) implicit-GEMM conv (WMMA, hi/lo split) + bias + normalize
    conv_wmma_kernel<<<HW / 128, 256, 0, stream>>>(input,       HW, W1h, W1l, W2h, W2l,
                                                   b1, b2, q1hat, q2hat,
                                                   (f16*)nullptr, (f16*)nullptr,
                                                   (f16*)nullptr, (f16*)nullptr);
    conv_wmma_kernel<<<S1 / 128, 256, 0, stream>>>(class_input, S1, W1h, W1l, W2h, W2l,
                                                   b1, b2, k1hat, k2hat,
                                                   kh1, kl1, kh2, kl2);
    conv_wmma_kernel<<<S2 / 128, 256, 0, stream>>>(saved_input, S2, W1h, W1l, W2h, W2l,
                                                   b1, b2, sk1hat, sk2hat,
                                                   skh1, skl1, skh2, skl2);

    // 3) per-class means of normalized k-vectors (deterministic single block)
    class_reduce_kernel<<<1, 256, 0, stream>>>(k1hat, k2hat, class_label, K1c, K2c);

    // 4) memory_map: split-f16 WMMA GEMM (S2 x S1, K=64) + 1x1 conv + sigmoid
    {
        dim3 grid(S1 / 128, S2 / 64);         // 32 x 64 blocks, 8 waves/block
        memory_map_wmma_kernel<<<grid, 256, 0, stream>>>(
            skh1, skl1, skh2, skl2, kh1, kl1, kh2, kl2, cc1w, cc1b, memory_map);
    }

    // 5) result_map: collapsed attention (per-class mean . qhat) + 1x1 + sigmoid
    result_map_kernel<<<HW / 256, 256, 0, stream>>>(
        q1hat, q2hat, K1c, K2c, cc2w, cc2b, result_map);
}
